// MultiSimilarityLossVectorized_72481868087462
// MI455X (gfx1250) — matrix operand order repaired and checked by reading, not compiled, not run on hardware
//
#include <hip/hip_runtime.h>
#include <hip/hip_bf16.h>

// Multi-Similarity loss, fused: sim = E E^T via bf16x3-split WMMA, masked
// exp row-sums in the epilogue, log1p + reduction in follow-up kernels.
// CDNA5 paths: v_wmma_f32_16x16x32_bf16, global_load_async_to_lds_b128
// (+ s_wait_asynccnt), ds_load_b128 fragment reads, global_prefetch.

typedef __attribute__((ext_vector_type(16))) __bf16 v16bf;
typedef __attribute__((ext_vector_type(8)))  float  v8f;

#define DIM       512
#define M_TILE    32    // rows per workgroup (2 waves x 16)
#define COL_SLICES 8

union FragBF {
    v16bf v;
    int4  c[2];
};

__device__ __forceinline__ unsigned short f32_to_bf16_rne(float x) {
    unsigned u = __float_as_uint(x);
    unsigned r = (u + 0x7FFFu + ((u >> 16) & 1u)) >> 16;
    return (unsigned short)r;
}

__device__ __forceinline__ float bf16_bits_to_f32(unsigned short h) {
    return __uint_as_float(((unsigned)h) << 16);
}

// Async global->LDS copy of 16 bytes (CDNA5 GLOBAL_LOAD_ASYNC_TO_LDS_B128,
// tracked by ASYNCcnt). ldsOff is the LDS byte address (low 32 bits of the
// generic pointer per the LDS aperture mapping).
__device__ __forceinline__ void async_copy_b128(unsigned ldsOff,
                                                const void* gptr) {
    unsigned long long ga = (unsigned long long)(size_t)gptr;
    asm volatile("global_load_async_to_lds_b128 %0, %1, off"
                 :: "v"(ldsOff), "v"(ga)
                 : "memory");
}

__device__ __forceinline__ void wait_asynccnt0() {
    asm volatile("s_wait_asynccnt 0x0" ::: "memory");
}

// ---------------------------------------------------------------------------
// Kernel 1: split fp32 embeddings into bf16 hi/lo planes; zero accumulators.
// ---------------------------------------------------------------------------
__global__ void ms_convert_kernel(const float* __restrict__ emb,
                                  unsigned short* __restrict__ ehi,
                                  unsigned short* __restrict__ elo,
                                  float* __restrict__ gpos,
                                  float* __restrict__ gneg,
                                  float* __restrict__ gscal,
                                  int total, int n) {
    int i = blockIdx.x * blockDim.x + threadIdx.x;
    if (i < total) {
        float x = emb[i];
        unsigned short h = f32_to_bf16_rne(x);
        float hf = bf16_bits_to_f32(h);
        float lo = x - hf;
        ehi[i] = h;
        elo[i] = f32_to_bf16_rne(lo);
    }
    if (i < n) { gpos[i] = 0.0f; gneg[i] = 0.0f; }
    if (i < 2) { gscal[i] = 0.0f; }
}

// ---------------------------------------------------------------------------
// Kernel 2: fused GEMM + masked-exp row sums.
// Block = 64 threads (2 waves). Each wave owns 16 rows; block owns 32 rows
// and one column slice of N/COL_SLICES columns. A strip staged in LDS via
// async copies.
// ---------------------------------------------------------------------------
__global__ __launch_bounds__(64)
void ms_gemm_kernel(const unsigned short* __restrict__ ehi,
                    const unsigned short* __restrict__ elo,
                    const int* __restrict__ labels,
                    float* __restrict__ gpos,
                    float* __restrict__ gneg,
                    int n, int colSlice) {
    __shared__ unsigned short sAhi[M_TILE * DIM];   // 32 KB
    __shared__ unsigned short sAlo[M_TILE * DIM];   // 32 KB

    const int lane = threadIdx.x & 31;
    const int wave = threadIdx.x >> 5;      // 0..1
    const int half = lane >> 4;             // 0/1 (wave half)
    const int l15  = lane & 15;
    const int m0   = blockIdx.x * M_TILE;
    const int colBase = blockIdx.y * colSlice;

    // ---- stage A strip (rows m0..m0+31, full K) into LDS via async DMA ----
    {
        const int nChunks = (M_TILE * DIM) / 8;     // 16B chunks (8 bf16 each)
        const unsigned short* srcHi = ehi + (size_t)m0 * DIM;
        const unsigned short* srcLo = elo + (size_t)m0 * DIM;
        const unsigned dHiBase = (unsigned)(size_t)(&sAhi[0]);
        const unsigned dLoBase = (unsigned)(size_t)(&sAlo[0]);
        for (int t = threadIdx.x; t < nChunks; t += blockDim.x) {
            async_copy_b128(dHiBase + (unsigned)t * 16u, srcHi + t * 8);
            async_copy_b128(dLoBase + (unsigned)t * 16u, srcLo + t * 8);
        }
        wait_asynccnt0();
    }
    __syncthreads();

    // ---- per-lane row metadata (C-tile layout: lanes 0-15 rows r, 16-31 rows r+8)
    int rowg[8];
    int labR[8];
#pragma unroll
    for (int r = 0; r < 8; ++r) {
        rowg[r] = m0 + wave * 16 + (half ? (r + 8) : r);
        labR[r] = labels[rowg[r]];
    }

    // A-fragment source row in LDS (A layout: both halves hold rows 0-15,
    // halves differ in K sub-range)
    const int aRow = wave * 16 + l15;
    const unsigned short* aHiRow = sAhi + aRow * DIM + half * 8;
    const unsigned short* aLoRow = sAlo + aRow * DIM + half * 8;

    float posAcc[8];
    float negAcc[8];
#pragma unroll
    for (int r = 0; r < 8; ++r) { posAcc[r] = 0.0f; negAcc[r] = 0.0f; }

    const int nColTiles = colSlice / 16;
    for (int nt = 0; nt < nColTiles; ++nt) {
        const int n0   = colBase + nt * 16;
        const int colg = n0 + l15;
        const int labC = labels[colg];

        // B layout: lane l15 holds column colg; V0-3 = K kb..kb+7,
        // V4-7 = K kb+16..kb+23 (kb = k0 + half*8), same as A.
        const unsigned short* bHiCol = ehi + (size_t)colg * DIM + half * 8;
        const unsigned short* bLoCol = elo + (size_t)colg * DIM + half * 8;

        // Prefetch next tile's B stream into cache (global_prefetch).
        if (nt + 1 < nColTiles) {
            const unsigned short* nHi = ehi + (size_t)(colg + 16) * DIM;
            const unsigned short* nLo = elo + (size_t)(colg + 16) * DIM;
            __builtin_prefetch(nHi, 0, 1);
            __builtin_prefetch(nLo, 0, 1);
        }

        v8f c = {};
#pragma unroll 4
        for (int k0 = 0; k0 < DIM; k0 += 32) {
            FragBF ahi, alo, bhi, blo;
            // A: V0-3 = K kb..kb+7, V4-7 = K kb+16..kb+23 (kb = k0 + half*8)
            ahi.c[0] = *(const int4*)(aHiRow + k0);
            ahi.c[1] = *(const int4*)(aHiRow + k0 + 16);
            alo.c[0] = *(const int4*)(aLoRow + k0);
            alo.c[1] = *(const int4*)(aLoRow + k0 + 16);
            // B: same K split as A
            bhi.c[0] = *(const int4*)(bHiCol + k0);
            bhi.c[1] = *(const int4*)(bHiCol + k0 + 16);
            blo.c[0] = *(const int4*)(bLoCol + k0);
            blo.c[1] = *(const int4*)(bLoCol + k0 + 16);

            // sim += Ahi*Bhi + Ahi*Blo + Alo*Bhi   (bf16x3 fp32 emulation)
            c = __builtin_amdgcn_wmma_f32_16x16x32_bf16(
                    false, ahi.v, false, bhi.v, (short)0, c, false, false);
            c = __builtin_amdgcn_wmma_f32_16x16x32_bf16(
                    false, ahi.v, false, blo.v, (short)0, c, false, false);
            c = __builtin_amdgcn_wmma_f32_16x16x32_bf16(
                    false, alo.v, false, bhi.v, (short)0, c, false, false);
        }

        // Epilogue: masked exp accumulation (branchless; single v_exp per
        // element since pos/neg masks are disjoint)
#pragma unroll
        for (int r = 0; r < 8; ++r) {
            float s    = c[r];
            bool  same = (labR[r] == labC);
            bool  diag = (rowg[r] == colg);
            float arg  = same ? (0.5f - s)            // exp(-alpha*(s-m)), alpha=1
                              : (10.0f * (s - 0.5f)); // exp(beta*(s-m)),  beta=10
            float e    = __expf(arg);
            posAcc[r] += (same && !diag) ? e : 0.0f;  // pos: same label, off-diag
            negAcc[r] += same ? 0.0f : e;             // neg: different label
        }
    }

    // ---- reduce across the 16 lanes sharing each row, then atomic-add ----
#pragma unroll
    for (int r = 0; r < 8; ++r) {
        float p = posAcc[r];
        float q = negAcc[r];
#pragma unroll
        for (int off = 8; off >= 1; off >>= 1) {
            p += __shfl_xor(p, off, 32);
            q += __shfl_xor(q, off, 32);
        }
        if (l15 == 0) {
            atomicAdd(&gpos[rowg[r]], p);
            atomicAdd(&gneg[rowg[r]], q);
        }
    }
}

// ---------------------------------------------------------------------------
// Kernel 3: per-row log1p terms + validity; block reduce; atomic total.
// ---------------------------------------------------------------------------
__global__ void ms_rows_kernel(const int* __restrict__ labels,
                               const float* __restrict__ gpos,
                               const float* __restrict__ gneg,
                               float* __restrict__ gscal,
                               int n) {
    int row = blockIdx.x * blockDim.x + threadIdx.x;
    float contrib = 0.0f;
    float validf  = 0.0f;
    if (row < n) {
        int lab = labels[row];
        int cnt = 0;
        for (int j = 0; j < n; ++j) cnt += (labels[j] == lab) ? 1 : 0;
        bool posAny = (cnt >= 2);          // another sample with same label
        bool negAny = ((n - cnt) >= 1);    // some sample with different label
        if (posAny && negAny) {
            contrib = log1pf(gpos[row]) + 0.1f * log1pf(gneg[row]);
            validf  = 1.0f;
        }
    }
    __shared__ float sC[256];
    __shared__ float sV[256];
    int t = threadIdx.x;
    sC[t] = contrib;
    sV[t] = validf;
    __syncthreads();
    for (int s = 128; s > 0; s >>= 1) {
        if (t < s) { sC[t] += sC[t + s]; sV[t] += sV[t + s]; }
        __syncthreads();
    }
    if (t == 0) {
        atomicAdd(&gscal[0], sC[0]);
        atomicAdd(&gscal[1], sV[0]);
    }
}

// ---------------------------------------------------------------------------
// Kernel 4: final scalar.
// ---------------------------------------------------------------------------
__global__ void ms_final_kernel(const float* __restrict__ gscal,
                                float* __restrict__ out) {
    float total  = gscal[0];
    float nvalid = gscal[1];
    out[0] = (nvalid > 0.0f) ? (total / nvalid) : 0.0f;
}

// ---------------------------------------------------------------------------
extern "C" void kernel_launch(void* const* d_in, const int* in_sizes, int n_in,
                              void* d_out, int out_size, void* d_ws, size_t ws_size,
                              hipStream_t stream) {
    const float* emb    = (const float*)d_in[0];
    const int*   labels = (const int*)d_in[1];
    float*       out    = (float*)d_out;

    const int n = in_sizes[1];           // 4096
    const int d = in_sizes[0] / n;       // 512 (== DIM)
    const int total = n * d;

    // workspace layout
    char* ws = (char*)d_ws;
    unsigned short* ehi = (unsigned short*)ws;                 // n*d bf16
    unsigned short* elo = ehi + (size_t)n * d;                 // n*d bf16
    float* gpos  = (float*)(elo + (size_t)n * d);              // n f32
    float* gneg  = gpos + n;                                   // n f32
    float* gscal = gneg + n;                                   // 2 f32 (total, nvalid)

    // 1) split to bf16 hi/lo + zero accumulators
    ms_convert_kernel<<<(total + 255) / 256, 256, 0, stream>>>(
        emb, ehi, elo, gpos, gneg, gscal, total, n);

    // 2) fused WMMA GEMM + masked exp row sums
    const int colSlice = n / COL_SLICES;                       // 512
    dim3 grid(n / M_TILE, COL_SLICES, 1);                      // 128 x 8
    ms_gemm_kernel<<<grid, 64, 0, stream>>>(ehi, elo, labels, gpos, gneg,
                                            n, colSlice);

    // 3) per-row terms + reduction
    ms_rows_kernel<<<(n + 255) / 256, 256, 0, stream>>>(labels, gpos, gneg,
                                                        gscal, n);

    // 4) final scalar
    ms_final_kernel<<<1, 1, 0, stream>>>(gscal, out);
}
